// CrossScaleAttention_64269890617678
// MI455X (gfx1250) — matrix operand order repaired and checked by reading, not compiled, not run on hardware
//
#include <hip/hip_runtime.h>
#include <hip/hip_bf16.h>

#define C      256
#define NHEADS 4
#define HDIM   64
#define BATCH  8
#define HW     48
#define NPIX   (HW * HW)          // 2304
#define PT_PER_B (NPIX / 16)      // 144
#define NKT    (NPIX / 32)        // 72 key tiles
#define LOG2E  1.44269504088896340736f

// padded LDS row strides (elements): +16B per row -> conflict-free b128 reads
#define QS_LD  72    // 64  + 8
#define KS_LD  72    // 64  + 8
#define VS_LD  40    // 32  + 8
#define PS_LD  40    // 32  + 8
#define XS_LD  264   // 256 + 8

typedef __attribute__((ext_vector_type(16))) __bf16 v16bf;
typedef __attribute__((ext_vector_type(8)))  __bf16 v8bf;
typedef __attribute__((ext_vector_type(8)))  float  v8f;

static __device__ __forceinline__ v8f wmma_bf16(v16bf a, v16bf b, v8f c) {
  return __builtin_amdgcn_wmma_f32_16x16x32_bf16(false, a, false, b, (short)0, c,
                                                 false, false);
}

static __device__ __forceinline__ v16bf ld2(const __bf16* p0, const __bf16* p1) {
  const v8bf lo = *(const v8bf*)p0;
  const v8bf hi = *(const v8bf*)p1;
  return __builtin_shufflevector(lo, hi, 0, 1, 2, 3, 4, 5, 6, 7,
                                 8, 9, 10, 11, 12, 13, 14, 15);
}

static __device__ __forceinline__ uint32_t lds_off(const void* p) {
  return (uint32_t)(uintptr_t)p;
}

// CDNA5 async global->LDS copy of 16 bytes (ASYNCcnt-tracked)
static __device__ __forceinline__ void async_cp16(const void* gptr, void* lptr) {
  asm volatile("global_load_async_to_lds_b128 %0, %1, off"
               :: "v"(lds_off(lptr)), "v"(gptr) : "memory");
}
static __device__ __forceinline__ void wait_async0() {
  asm volatile("s_wait_asynccnt 0" ::: "memory");
}

// ---------------------------------------------------------------------------
// Kernel 0: one-time weight conversion f32 -> bf16 row-major (L2-resident)
// grid = (C*C/(4*256), 4), block = 256
// ---------------------------------------------------------------------------
__global__ __launch_bounds__(256)
void wconv_kernel(const float* __restrict__ Wq, const float* __restrict__ Wk,
                  const float* __restrict__ Wv, const float* __restrict__ Wo,
                  __bf16* __restrict__ Wb)
{
  const float* src = (blockIdx.y == 0) ? Wq : (blockIdx.y == 1) ? Wk
                   : (blockIdx.y == 2) ? Wv : Wo;
  __bf16* dst = Wb + (size_t)blockIdx.y * C * C;
  const int i = (blockIdx.x * 256 + threadIdx.x) * 4;
  const float4 f = *(const float4*)(src + i);
  dst[i + 0] = (__bf16)f.x;
  dst[i + 1] = (__bf16)f.y;
  dst[i + 2] = (__bf16)f.z;
  dst[i + 3] = (__bf16)f.w;
}

// ---------------------------------------------------------------------------
// Kernel 1: QKV projections. GEMM [C,C] x [C,N] per batch per projection.
// grid = (BATCH*PT_PER_B, 2, 3), block = 256 (8 waves). Wave -> 16x16 tile.
// Q,K written transposed [b*h][n][64]; V channel-major [b][C][N].
// ---------------------------------------------------------------------------
__global__ __launch_bounds__(256)
void qkv_proj_kernel(const float* __restrict__ x, const __bf16* __restrict__ Wb,
                     const float* __restrict__ bq, const float* __restrict__ bk,
                     const float* __restrict__ bv,
                     __bf16* __restrict__ Qw, __bf16* __restrict__ Kw,
                     __bf16* __restrict__ Vw)
{
  __shared__ __attribute__((aligned(16))) __bf16 Xs[16 * XS_LD]; // [p][c] padded

  const int proj = blockIdx.z;
  const __bf16* Wm  = Wb + (size_t)proj * C * C;
  const float* bias = (proj == 0) ? bq : (proj == 1) ? bk : bv;
  __bf16*      out  = (proj == 0) ? Qw : (proj == 1) ? Kw : Vw;

  const int tid  = threadIdx.x;
  const int wave = tid >> 5;
  const int lane = tid & 31;
  const int half = lane >> 4;
  const int r    = lane & 15;

  const int mo = (blockIdx.y * 8 + wave) * 16;
  const int pt = blockIdx.x;
  const int b  = pt / PT_PER_B;
  const int p0 = (pt % PT_PER_B) * 16;

  const float* xb = x + (size_t)b * C * NPIX;

  // stage x tile [16 p][256 c] as bf16 (transpose + convert)
  for (int i = tid; i < 16 * 256; i += 256) {
    const int c = i >> 4, p = i & 15;
    Xs[p * XS_LD + c] = (__bf16)xb[(size_t)c * NPIX + p0 + p];
  }
  __syncthreads();

  v8f acc;
#pragma unroll
  for (int g = 0; g < 8; ++g) acc[g] = bias[mo + g + half * 8];

#pragma unroll
  for (int c0 = 0; c0 < C; c0 += 32) {
    const __bf16* wrow = Wm + (size_t)(mo + r) * C + c0 + half * 8;
    const __bf16* xc   = &Xs[r * XS_LD + c0 + half * 8];
    acc = wmma_bf16(ld2(wrow, wrow + 16), ld2(xc, xc + 16), acc);
  }

  if (proj < 2) {
    const int h  = mo >> 6;
    const int d0 = (mo & 63) + half * 8;
    __bf16* ot = out + ((size_t)(b * NHEADS + h) * NPIX + p0 + r) * HDIM + d0;
#pragma unroll
    for (int g = 0; g < 8; ++g) ot[g] = (__bf16)acc[g];
  } else {
    __bf16* ob = out + (size_t)b * C * NPIX;
#pragma unroll
    for (int g = 0; g < 8; ++g)
      ob[(size_t)(mo + g + half * 8) * NPIX + p0 + r] = (__bf16)acc[g];
  }
}

// ---------------------------------------------------------------------------
// Kernel 2: fused flash attention per (batch, head).
// grid = (NPIX/128, BATCH*NHEADS), block = 256 (8 waves). Wave owns 16 q rows.
// Q/K/V staged with global_load_async_to_lds_b128 (K/V double-buffered),
// padded LDS rows -> conflict-free ds_load_b128 fragments.
// ---------------------------------------------------------------------------
__global__ __launch_bounds__(256)
void flash_attn_kernel(const __bf16* __restrict__ Qt,
                       const __bf16* __restrict__ Kt,
                       const __bf16* __restrict__ Vw,
                       __bf16* __restrict__ AO)
{
  __shared__ __attribute__((aligned(16))) __bf16 Qs[128 * QS_LD];
  __shared__ __attribute__((aligned(16))) __bf16 Ks[2][32 * KS_LD];
  __shared__ __attribute__((aligned(16))) __bf16 Vs[2][64 * VS_LD];
  __shared__ __attribute__((aligned(16))) __bf16 Ps[8][16 * PS_LD];

  const int bh    = blockIdx.y;
  const int b     = bh >> 2;
  const int h     = bh & 3;
  const int qbase = blockIdx.x * 128;
  const int tid   = threadIdx.x;
  const int wave  = tid >> 5;
  const int lane  = tid & 31;
  const int half  = lane >> 4;
  const int r     = lane & 15;

  const __bf16* Qb = Qt + (size_t)(b * NHEADS + h) * NPIX * HDIM;
  const __bf16* Kb = Kt + (size_t)(b * NHEADS + h) * NPIX * HDIM;
  const __bf16* Vb = Vw + ((size_t)b * C + h * HDIM) * NPIX;       // [d][N]

  // async-stage Q tile [128][64] -> padded rows (8 chunks of 16B per row)
  {
    const __bf16* g = Qb + (size_t)qbase * HDIM;
#pragma unroll
    for (int j = 0; j < 4; ++j) {
      const int i   = tid + j * 256;       // chunk id, 1024 total
      const int row = i >> 3, off = (i & 7) * 8;
      async_cp16(g + (size_t)i * 8, &Qs[row * QS_LD + off]);
    }
  }
  // async-stage K/V tile 0
  {
    const int krow = tid >> 3, koff = (tid & 7) * 8;
    async_cp16(Kb + (size_t)tid * 8, &Ks[0][krow * KS_LD + koff]);
    const int d = tid >> 2, kk = (tid & 3) * 8;
    async_cp16(Vb + (size_t)d * NPIX + kk, &Vs[0][d * VS_LD + kk]);
  }
  wait_async0();
  __syncthreads();

  const int qr0 = wave * 16;
  const int q   = qr0 + r;
  v8f accO[4];
  float mrow[8], lrow[8];
#pragma unroll
  for (int nd = 0; nd < 4; ++nd)
#pragma unroll
    for (int g = 0; g < 8; ++g) accO[nd][g] = 0.0f;
#pragma unroll
  for (int g = 0; g < 8; ++g) { mrow[g] = -1e30f; lrow[g] = 0.0f; }

  const float SC = 0.125f * LOG2E;

  for (int t = 0; t < NKT; ++t) {
    const int cur = t & 1;

    // prefetch next K/V tile (overlaps the WMMAs below)
    if (t + 1 < NKT) {
      const int k1 = (t + 1) * 32;
      const int krow = tid >> 3, koff = (tid & 7) * 8;
      async_cp16(Kb + (size_t)k1 * HDIM + tid * 8,
                 &Ks[cur ^ 1][krow * KS_LD + koff]);
      const int d = tid >> 2, kk = (tid & 3) * 8;
      async_cp16(Vb + (size_t)d * NPIX + k1 + kk,
                 &Vs[cur ^ 1][d * VS_LD + kk]);
    }

    // A fragments of Q^T (row q, two d-chunks of 32), reused by both halves
    v16bf aq[2];
#pragma unroll
    for (int dc = 0; dc < 2; ++dc) {
      const __bf16* qp = &Qs[q * QS_LD + dc * 32 + half * 8];
      aq[dc] = ld2(qp, qp + 16);
    }

    // S[16q x 32k] in two 16-col halves
    v8f s[2];
#pragma unroll
    for (int kh = 0; kh < 2; ++kh) {
      v8f sa;
#pragma unroll
      for (int g = 0; g < 8; ++g) sa[g] = 0.0f;
#pragma unroll
      for (int dc = 0; dc < 2; ++dc) {
        const __bf16* kp = &Ks[cur][(kh * 16 + r) * KS_LD + dc * 32 + half * 8];
        sa = wmma_bf16(aq[dc], ld2(kp, kp + 16), sa);
      }
      s[kh] = sa;
    }

    // online softmax
    __bf16* Pw = &Ps[wave][0];
#pragma unroll
    for (int g = 0; g < 8; ++g) {
      const float s0 = s[0][g] * SC;
      const float s1 = s[1][g] * SC;
      float mx = fmaxf(s0, s1);
#pragma unroll
      for (int off = 1; off < 16; off <<= 1) mx = fmaxf(mx, __shfl_xor(mx, off));
      const float mnew = fmaxf(mrow[g], mx);
      const float p0 = exp2f(s0 - mnew);
      const float p1 = exp2f(s1 - mnew);
      float ps = p0 + p1;
#pragma unroll
      for (int off = 1; off < 16; off <<= 1) ps += __shfl_xor(ps, off);
      const float alpha = exp2f(mrow[g] - mnew);
      lrow[g] = lrow[g] * alpha + ps;
      mrow[g] = mnew;
#pragma unroll
      for (int nd = 0; nd < 4; ++nd) accO[nd][g] *= alpha;
      const int prow = (g + half * 8) * PS_LD;
      Pw[prow + r]      = (__bf16)p0;
      Pw[prow + 16 + r] = (__bf16)p1;
    }
    asm volatile("" ::: "memory");

    // O[16q x 64d] += P @ V^T
    const __bf16* pr = Pw + r * PS_LD + half * 8;
    const v16bf a2 = ld2(pr, pr + 16);
#pragma unroll
    for (int nd = 0; nd < 4; ++nd) {
      const __bf16* vp = &Vs[cur][(nd * 16 + r) * VS_LD + half * 8];
      accO[nd] = wmma_bf16(a2, ld2(vp, vp + 16), accO[nd]);
    }

    wait_async0();
    __syncthreads();
  }

  // finalize: /l, write AO in [b][n][C]
  __bf16* AOb = AO + (size_t)b * NPIX * C;
#pragma unroll
  for (int g = 0; g < 8; ++g) {
    const float inv = 1.0f / lrow[g];
    const size_t row = (size_t)(qbase + qr0 + g + half * 8) * C + h * HDIM;
#pragma unroll
    for (int nd = 0; nd < 4; ++nd)
      AOb[row + nd * 16 + r] = (__bf16)(accO[nd][g] * inv);
  }
}

// ---------------------------------------------------------------------------
// Kernel 3: output projection + gamma residual, fp32 out.
// grid = (BATCH*PT_PER_B, 2), block = 256. AO tile async-staged with padded
// LDS rows (row = 32 chunks of 16B).
// ---------------------------------------------------------------------------
__global__ __launch_bounds__(256)
void out_proj_kernel(const float* __restrict__ x, const __bf16* __restrict__ Wob,
                     const float* __restrict__ bo, const float* __restrict__ gamma,
                     const __bf16* __restrict__ AO, float* __restrict__ out)
{
  __shared__ __attribute__((aligned(16))) __bf16 AOs[16 * XS_LD]; // [p][c] padded

  const int tid  = threadIdx.x;
  const int wave = tid >> 5;
  const int lane = tid & 31;
  const int half = lane >> 4;
  const int r    = lane & 15;

  const int mo = (blockIdx.y * 8 + wave) * 16;
  const int pt = blockIdx.x;
  const int b  = pt / PT_PER_B;
  const int p0 = (pt % PT_PER_B) * 16;

  // async-stage AO tile [16][256] (contiguous source, padded LDS rows)
  {
    const __bf16* g = AO + ((size_t)b * NPIX + p0) * C;
#pragma unroll
    for (int j = 0; j < 2; ++j) {
      const int i = tid + j * 256;         // chunk id, 512 total
      const int p = i >> 5, cc = (i & 31) * 8;
      async_cp16(g + (size_t)i * 8, &AOs[p * XS_LD + cc]);
    }
  }
  wait_async0();
  __syncthreads();

  v8f acc;
#pragma unroll
  for (int g = 0; g < 8; ++g) acc[g] = bo[mo + g + half * 8];

#pragma unroll
  for (int c0 = 0; c0 < C; c0 += 32) {
    const __bf16* wrow = Wob + (size_t)(mo + r) * C + c0 + half * 8;
    const __bf16* ac   = &AOs[r * XS_LD + c0 + half * 8];
    acc = wmma_bf16(ld2(wrow, wrow + 16), ld2(ac, ac + 16), acc);
  }

  const float g0 = gamma[0];
  const float* xb = x   + (size_t)b * C * NPIX;
  float*       ob = out + (size_t)b * C * NPIX;
#pragma unroll
  for (int g = 0; g < 8; ++g) {
    const size_t idx = (size_t)(mo + g + half * 8) * NPIX + p0 + r;
    ob[idx] = xb[idx] + g0 * acc[g];
  }
}

// ---------------------------------------------------------------------------
extern "C" void kernel_launch(void* const* d_in, const int* in_sizes, int n_in,
                              void* d_out, int out_size, void* d_ws, size_t ws_size,
                              hipStream_t stream)
{
  const float* x     = (const float*)d_in[0];
  const float* Wq    = (const float*)d_in[1];
  const float* bq    = (const float*)d_in[2];
  const float* Wk    = (const float*)d_in[3];
  const float* bk    = (const float*)d_in[4];
  const float* Wv    = (const float*)d_in[5];
  const float* bv    = (const float*)d_in[6];
  const float* Wo    = (const float*)d_in[7];
  const float* bo    = (const float*)d_in[8];
  const float* gamma = (const float*)d_in[9];
  float* out = (float*)d_out;

  const size_t elems = (size_t)BATCH * C * NPIX;
  __bf16* Qw = (__bf16*)d_ws;        // [b*h][n][64]
  __bf16* Kw = Qw + elems;           // [b*h][n][64]
  __bf16* Vw = Kw + elems;           // [b][C][N]
  __bf16* AO = Vw + elems;           // [b][n][C]
  __bf16* Wb = AO + elems;           // 4 x [C][C] bf16 (q,k,v,o)

  const dim3 blk(256);

  wconv_kernel<<<dim3(C * C / (4 * 256), 4), blk, 0, stream>>>(
      Wq, Wk, Wv, Wo, Wb);

  qkv_proj_kernel<<<dim3(BATCH * PT_PER_B, 2, 3), blk, 0, stream>>>(
      x, Wb, bq, bk, bv, Qw, Kw, Vw);

  flash_attn_kernel<<<dim3(NPIX / 128, BATCH * NHEADS), blk, 0, stream>>>(
      Qw, Kw, Vw, AO);

  out_proj_kernel<<<dim3(BATCH * PT_PER_B, 2), blk, 0, stream>>>(
      x, Wb + 3 * C * C, bo, gamma, AO, out);
}
